// Decoder_16398185136237
// MI455X (gfx1250) — compile-verified, matrix operand order checked by reading, not run on hardware
//
#include <hip/hip_runtime.h>
#include <hip/hip_bf16.h>
#include <math.h>

// Problem constants (from reference): B=1024, T-1=127, E=D=256, F=1
#define NB   1024
#define NT   127
#define NE   256
#define ND   256

typedef __bf16 bf16_t;
typedef __attribute__((ext_vector_type(16))) __bf16 v16bf;
typedef __attribute__((ext_vector_type(4)))  __bf16 v4bf;
typedef __attribute__((ext_vector_type(8)))  float  v8f;

// ---------------------------------------------------------------------------
// WMMA fragment loader: 16(row) x 32(col) bf16 tile from a row-major matrix.
// CDNA5 16-bit A layout (ISA 7.12.2): lane<16 -> row=lane, K={0..7,16..23};
// lane>=16 -> row=lane-16, K={8..15,24..31}. B (=W[n][k], rows over N) uses
// the transposed-symmetric layout, so the same loader serves both for X@W^T.
// ---------------------------------------------------------------------------
static __device__ __forceinline__ v16bf load_frag16x32(const bf16_t* __restrict__ base, int ld) {
  const int lane = threadIdx.x & 31;
  const int r    = lane & 15;
  const int half = lane >> 4;
  const bf16_t* p = base + (size_t)r * ld + half * 8;
  v16bf f;
#pragma unroll
  for (int i = 0; i < 8; ++i) f[i] = p[i];
#pragma unroll
  for (int i = 0; i < 8; ++i) f[8 + i] = p[16 + i];
  return f;
}

static __device__ __forceinline__ float wave_sum(float v) {
#pragma unroll
  for (int m = 16; m >= 1; m >>= 1) v += __shfl_xor(v, m, 32);
  return v;
}
static __device__ __forceinline__ float wave_max(float v) {
#pragma unroll
  for (int m = 16; m >= 1; m >>= 1) v = fmaxf(v, __shfl_xor(v, m, 32));
  return v;
}

// ---------------------------------------------------------------------------
// Init: split/convert W1 -> bf16 (h-part / enc-part), W_hh -> bf16, zero h0.
// ---------------------------------------------------------------------------
__global__ void k_init(const float* __restrict__ W1, const float* __restrict__ Whh,
                       bf16_t* __restrict__ W1h, bf16_t* __restrict__ W1e,
                       bf16_t* __restrict__ Whh_bf,
                       float* __restrict__ h0f, bf16_t* __restrict__ h0b) {
  const int i = blockIdx.x * blockDim.x + threadIdx.x;   // 0 .. 262143
  if (i < NE * (ND + NE)) {                              // W1 is [256 x 512]
    const float v = W1[i];
    const int e = i >> 9, c = i & 511;
    if (c < ND) W1h[e * ND + c] = (bf16_t)v;
    else        W1e[e * NE + (c - ND)] = (bf16_t)v;
  }
  if (i < 3 * ND * ND) Whh_bf[i] = (bf16_t)Whh[i];       // [768 x 256]
  h0f[i] = 0.0f;                                         // i < 1024*256 always
  h0b[i] = (bf16_t)0.0f;
}

// enc_bf = bf16(input_encoded), 4 elems / thread
__global__ void k_cast_enc(const float* __restrict__ in, bf16_t* __restrict__ out, long n4) {
  const long i = (long)blockIdx.x * blockDim.x + threadIdx.x;
  if (i >= n4) return;
  const float4 v = ((const float4*)in)[i];
  v4bf o; o[0] = (bf16_t)v.x; o[1] = (bf16_t)v.y; o[2] = (bf16_t)v.z; o[3] = (bf16_t)v.w;
  ((v4bf*)out)[i] = o;
}

// ---------------------------------------------------------------------------
// encproj[m][e] = sum_k enc_bf[m][k] * W1e[e][k] + b1[e]   (m = b*127+t)
// M=130048, N=256, K=256.  One wave per 16x16 output tile, 8 wmma per wave.
// ---------------------------------------------------------------------------
__global__ void k_encproj(const bf16_t* __restrict__ A, const bf16_t* __restrict__ W1e,
                          const float* __restrict__ b1, bf16_t* __restrict__ out) {
  const int wid  = threadIdx.x >> 5;
  const int lane = threadIdx.x & 31;
  const int m0 = (blockIdx.x * 8 + wid) * 16;
  const int n0 = blockIdx.y * 16;
  v8f acc = {};
#pragma unroll
  for (int k = 0; k < NE; k += 32) {
    v16bf a = load_frag16x32(A   + (size_t)m0 * NE + k, NE);
    v16bf b = load_frag16x32(W1e + (size_t)n0 * NE + k, NE);
    acc = __builtin_amdgcn_wmma_f32_16x16x32_bf16(false, a, false, b, (short)0, acc, false, false);
  }
  const int half = lane >> 4;
  const int n = n0 + (lane & 15);
  const float bias = b1[n];
#pragma unroll
  for (int v = 0; v < 8; ++v) {
    const int m = m0 + v + 8 * half;
    out[(size_t)m * NE + n] = (bf16_t)(acc[v] + bias);
  }
}

// hproj[b][e] = sum_d h_bf[b][d] * W1h[e][d]   (M=1024, N=256, K=256), f32 out
__global__ void k_hproj(const bf16_t* __restrict__ h_bf, const bf16_t* __restrict__ W1h,
                        float* __restrict__ hproj) {
  const int wid  = threadIdx.x >> 5;
  const int lane = threadIdx.x & 31;
  const int m0 = (blockIdx.x * 8 + wid) * 16;
  const int n0 = blockIdx.y * 16;
  v8f acc = {};
#pragma unroll
  for (int k = 0; k < ND; k += 32) {
    v16bf a = load_frag16x32(h_bf + (size_t)m0 * ND + k, ND);
    v16bf b = load_frag16x32(W1h + (size_t)n0 * ND + k, ND);
    acc = __builtin_amdgcn_wmma_f32_16x16x32_bf16(false, a, false, b, (short)0, acc, false, false);
  }
  const int half = lane >> 4;
  const int n = n0 + (lane & 15);
#pragma unroll
  for (int v = 0; v < 8; ++v)
    hproj[(size_t)(m0 + v + 8 * half) * NE + n] = acc[v];
}

// ---------------------------------------------------------------------------
// Fused attention for one timestep: one workgroup (256 thr) per batch row b.
//   scores[t] = W2 . tanh(hproj[b] + encproj[b,t]) + b2   (t = 0..126)
//   a = softmax(scores);  ctx[b] = sum_t a[t]*enc[b,t];   y_tilde[b] = fc(...)
// encproj + enc_bf sweeps are bf16 and L2-resident (133 MB < 192 MB L2).
// ---------------------------------------------------------------------------
__global__ void k_attn(const float* __restrict__ hproj, const bf16_t* __restrict__ encproj,
                       const bf16_t* __restrict__ enc_bf,
                       const float* __restrict__ W2, const float* __restrict__ b2,
                       const float* __restrict__ Wfc, const float* __restrict__ bfc,
                       const float* __restrict__ y_hist, int t,
                       float* __restrict__ ctx, float* __restrict__ y_tilde) {
  __shared__ float sh_h[256];
  __shared__ float sh_w2[256];
  __shared__ float sh_sc[128];
  __shared__ float sh_red[2];
  const int tid = threadIdx.x;
  const int b   = blockIdx.x;
  sh_h[tid]  = hproj[(size_t)b * NE + tid];
  sh_w2[tid] = W2[tid];
  __syncthreads();
  const int wid = tid >> 5, lane = tid & 31;
  const float b2v = b2[0];
  for (int tt = wid; tt < NT; tt += 8) {
    const bf16_t* ep = encproj + ((size_t)b * NT + tt) * NE;
    float p = 0.0f;
    for (int e = lane; e < NE; e += 32)
      p += sh_w2[e] * tanhf(sh_h[e] + (float)ep[e]);
    p = wave_sum(p);
    if (lane == 0) sh_sc[tt] = p + b2v;
  }
  __syncthreads();
  if (wid == 0) {
    float mx = -3.0e38f;
    for (int tt = lane; tt < NT; tt += 32) mx = fmaxf(mx, sh_sc[tt]);
    mx = wave_max(mx);
    if (lane == 0) sh_red[0] = mx;
  }
  __syncthreads();
  const float mx = sh_red[0];
  if (tid < NT) sh_sc[tid] = __expf(sh_sc[tid] - mx);
  __syncthreads();
  if (wid == 0) {
    float s = 0.0f;
    for (int tt = lane; tt < NT; tt += 32) s += sh_sc[tt];
    s = wave_sum(s);
    if (lane == 0) sh_red[1] = 1.0f / s;
  }
  __syncthreads();
  const float inv = sh_red[1];
  // context: thread tid owns column e=tid; per-t reads are fully coalesced
  const bf16_t* ebase = enc_bf + (size_t)b * NT * NE + tid;
  float acc = 0.0f;
  for (int tt = 0; tt < NT; ++tt)
    acc += sh_sc[tt] * (float)ebase[(size_t)tt * NE];
  acc *= inv;
  ctx[(size_t)b * NE + tid] = acc;
  // y_tilde[b] = Wfc[0:256].ctx + Wfc[256]*y_t + bfc
  __syncthreads();
  sh_h[tid] = Wfc[tid] * acc;
  __syncthreads();
  for (int s = 128; s >= 1; s >>= 1) {
    if (tid < s) sh_h[tid] += sh_h[tid + s];
    __syncthreads();
  }
  if (tid == 0)
    y_tilde[b] = sh_h[0] + Wfc[NE] * y_hist[(size_t)b * NT + t] + bfc[0];
}

// ---------------------------------------------------------------------------
// Fused GRU step: gh = h @ W_hh^T (3 gate tiles per wave, shared A fragment),
// then gate math in the epilogue.  Double-buffered h (read cur, write nxt).
// gi is rank-1 (F=1): gi[:,g*D+d] = y_tilde[b]*W_ih[g*D+d] + b_ih[g*D+d].
// ---------------------------------------------------------------------------
__global__ void k_gru(const bf16_t* __restrict__ h_bf, const float* __restrict__ h_f,
                      const bf16_t* __restrict__ Whh_bf,
                      const float* __restrict__ W_ih, const float* __restrict__ b_ih,
                      const float* __restrict__ b_hh, const float* __restrict__ y_tilde,
                      float* __restrict__ h_f_out, bf16_t* __restrict__ h_bf_out) {
  const int wid  = threadIdx.x >> 5;
  const int lane = threadIdx.x & 31;
  const int b0 = (blockIdx.x * 8 + wid) * 16;
  const int d0 = blockIdx.y * 16;
  v8f ar = {}, az = {}, an = {};
#pragma unroll
  for (int k = 0; k < ND; k += 32) {
    v16bf a  = load_frag16x32(h_bf   + (size_t)b0 * ND + k, ND);
    v16bf br = load_frag16x32(Whh_bf + (size_t)(d0          ) * ND + k, ND);
    v16bf bz = load_frag16x32(Whh_bf + (size_t)(ND + d0     ) * ND + k, ND);
    v16bf bn = load_frag16x32(Whh_bf + (size_t)(2 * ND + d0 ) * ND + k, ND);
    ar = __builtin_amdgcn_wmma_f32_16x16x32_bf16(false, a, false, br, (short)0, ar, false, false);
    az = __builtin_amdgcn_wmma_f32_16x16x32_bf16(false, a, false, bz, (short)0, az, false, false);
    an = __builtin_amdgcn_wmma_f32_16x16x32_bf16(false, a, false, bn, (short)0, an, false, false);
  }
  const int half = lane >> 4;
  const int d = d0 + (lane & 15);
  const float wir = W_ih[d], wiz = W_ih[ND + d], win = W_ih[2 * ND + d];
  const float bir = b_ih[d], biz = b_ih[ND + d], binn = b_ih[2 * ND + d];
  const float bhr = b_hh[d], bhz = b_hh[ND + d], bhn  = b_hh[2 * ND + d];
#pragma unroll
  for (int v = 0; v < 8; ++v) {
    const int b = b0 + v + 8 * half;
    const float yt = y_tilde[b];
    const float r = 1.0f / (1.0f + __expf(-(yt * wir + bir + ar[v] + bhr)));
    const float z = 1.0f / (1.0f + __expf(-(yt * wiz + biz + az[v] + bhz)));
    const float n = tanhf(yt * win + binn + r * (an[v] + bhn));
    const float hold = h_f[(size_t)b * ND + d];
    const float hn = (1.0f - z) * n + z * hold;
    h_f_out [(size_t)b * ND + d] = hn;
    h_bf_out[(size_t)b * ND + d] = (bf16_t)hn;
  }
}

// out[b] = Wfin[0:256].h[b] + Wfin[256:512].ctx[b] + bfin
__global__ void k_out(const float* __restrict__ h, const float* __restrict__ ctx,
                      const float* __restrict__ Wfin, const float* __restrict__ bfin,
                      float* __restrict__ out) {
  __shared__ float sh[256];
  const int tid = threadIdx.x, b = blockIdx.x;
  sh[tid] = Wfin[tid] * h[(size_t)b * ND + tid] + Wfin[ND + tid] * ctx[(size_t)b * NE + tid];
  __syncthreads();
  for (int s = 128; s >= 1; s >>= 1) {
    if (tid < s) sh[tid] += sh[tid + s];
    __syncthreads();
  }
  if (tid == 0) out[b] = sh[0] + bfin[0];
}

// ---------------------------------------------------------------------------
extern "C" void kernel_launch(void* const* d_in, const int* in_sizes, int n_in,
                              void* d_out, int out_size, void* d_ws, size_t ws_size,
                              hipStream_t stream) {
  (void)in_sizes; (void)n_in; (void)out_size; (void)ws_size;
  const float* input_encoded = (const float*)d_in[0];
  const float* y_history     = (const float*)d_in[1];
  const float* W1   = (const float*)d_in[2];
  const float* b1   = (const float*)d_in[3];
  const float* W2   = (const float*)d_in[4];
  const float* b2   = (const float*)d_in[5];
  const float* W_ih = (const float*)d_in[6];
  const float* Whh  = (const float*)d_in[7];
  const float* b_ih = (const float*)d_in[8];
  const float* b_hh = (const float*)d_in[9];
  const float* Wfc  = (const float*)d_in[10];
  const float* bfc  = (const float*)d_in[11];
  const float* Wfin = (const float*)d_in[12];
  const float* bfin = (const float*)d_in[13];
  float* out = (float*)d_out;

  // workspace carve-up (aligned to 256B); total ~138 MB
  char* ws = (char*)d_ws;
  auto carve = [&](size_t bytes) -> char* {
    char* p = ws; ws += (bytes + 255) & ~(size_t)255; return p;
  };
  const size_t enc_elems = (size_t)NB * NT * NE;           // 33,292,288
  bf16_t* enc_bf   = (bf16_t*)carve(enc_elems * 2);        // 66.6 MB (L2-resident)
  bf16_t* encproj  = (bf16_t*)carve(enc_elems * 2);        // 66.6 MB (L2-resident)
  bf16_t* W1h_bf   = (bf16_t*)carve((size_t)NE * ND * 2);
  bf16_t* W1e_bf   = (bf16_t*)carve((size_t)NE * NE * 2);
  bf16_t* Whh_bf   = (bf16_t*)carve((size_t)3 * ND * ND * 2);
  float*  h_f[2]   = {(float*)carve((size_t)NB * ND * 4), (float*)carve((size_t)NB * ND * 4)};
  bf16_t* h_b[2]   = {(bf16_t*)carve((size_t)NB * ND * 2), (bf16_t*)carve((size_t)NB * ND * 2)};
  float*  hproj    = (float*)carve((size_t)NB * NE * 4);
  float*  ctx      = (float*)carve((size_t)NB * NE * 4);
  float*  y_tilde  = (float*)carve((size_t)NB * 4);

  // one-time prep
  k_init<<<dim3((NB * ND) / 256), 256, 0, stream>>>(W1, Whh, W1h_bf, W1e_bf, Whh_bf, h_f[0], h_b[0]);
  const long n4 = (long)(enc_elems / 4);
  k_cast_enc<<<dim3((unsigned)((n4 + 255) / 256)), 256, 0, stream>>>(input_encoded, enc_bf, n4);
  // encproj GEMM: M=130048 -> 1016 blocks of 8 m-tiles; N=256 -> 16 n-tiles
  k_encproj<<<dim3((NB * NT) / 128, NE / 16), 256, 0, stream>>>(enc_bf, W1e_bf, b1, encproj);

  // 127 sequential recurrence steps
  for (int t = 0; t < NT; ++t) {
    const int cur = t & 1, nxt = cur ^ 1;
    k_hproj<<<dim3(NB / 128, NE / 16), 256, 0, stream>>>(h_b[cur], W1h_bf, hproj);
    k_attn<<<dim3(NB), 256, 0, stream>>>(hproj, encproj, enc_bf, W2, b2, Wfc, bfc,
                                         y_history, t, ctx, y_tilde);
    k_gru<<<dim3(NB / 128, ND / 16), 256, 0, stream>>>(h_b[cur], h_f[cur], Whh_bf,
                                                       W_ih, b_ih, b_hh, y_tilde,
                                                       h_f[nxt], h_b[nxt]);
  }
  // final h lives in buffer (127 & 1) ^ 0 ... written at t=126 into nxt=(126&1)^1=1
  k_out<<<dim3(NB), 256, 0, stream>>>(h_f[1], ctx, Wfin, bfin, out);
}
